// QuantizedBlock_30623116820575
// MI455X (gfx1250) — 16-bit and 8-bit matrix operand order repaired and checked by reading, compiled, not run on hardware
//
#include <hip/hip_runtime.h>

typedef __attribute__((ext_vector_type(8)))  int       v8i;
typedef __attribute__((ext_vector_type(8)))  float     v8f;
typedef __attribute__((ext_vector_type(16))) _Float16  v16h;
typedef _Float16 half_t;

#define BDIM   32
#define NSEQ   577
#define CDIM   1024
#define NHEAD  16
#define DHEAD  64
#define HIDDEN 4096
#define MROWS  (BDIM * NSEQ)
#define QTILES 37
#define SPAD   608
#define NPAD   640
#define LROW   20

union U8  { v8i  i; unsigned u[8]; };
union U16 { v16h h; unsigned u[8]; };

__global__ __launch_bounds__(256) void k_zerobuf(uint4* p, size_t n16) {
  size_t i = (size_t)blockIdx.x * blockDim.x + threadIdx.x;
  size_t stride = (size_t)gridDim.x * blockDim.x;
  uint4 z = {0u, 0u, 0u, 0u};
  for (; i < n16; i += stride) p[i] = z;
}

__global__ __launch_bounds__(256) void k_zero4(unsigned* p) {
  if (threadIdx.x < 4) p[threadIdx.x] = 0u;
}

__global__ __launch_bounds__(256) void k_absmax_tanh(const float* __restrict__ w,
                                                     int n, unsigned* out) {
  float m = 0.f;
  for (int i = blockIdx.x * blockDim.x + threadIdx.x; i < n; i += gridDim.x * blockDim.x)
    m = fmaxf(m, fabsf(tanhf(w[i])));
  for (int off = 16; off; off >>= 1) m = fmaxf(m, __shfl_xor(m, off, 32));
  if ((threadIdx.x & 31) == 0) atomicMax(out, __float_as_uint(m));
}

__global__ __launch_bounds__(256) void k_wquant(const float* __restrict__ w, int n,
                                                const unsigned* mb, signed char* __restrict__ out) {
  float inv = 1.f / __uint_as_float(*mb);
  for (int i = blockIdx.x * blockDim.x + threadIdx.x; i < n; i += gridDim.x * blockDim.x) {
    float wn = tanhf(w[i]) * inv;
    int q = 2 * (int)rintf((0.5f * wn + 0.5f) * 15.f) - 15;
    out[i] = (signed char)q;
  }
}

__global__ __launch_bounds__(256) void k_ln_quant(const float* __restrict__ x,
                                                  const float* __restrict__ g,
                                                  const float* __restrict__ bt,
                                                  unsigned char* __restrict__ out) {
  int row = blockIdx.x, t = threadIdx.x;
  const float* xr = x + (size_t)row * CDIM;
  float vals[4], s = 0.f, sq = 0.f;
#pragma unroll
  for (int i = 0; i < 4; ++i) {
    float v = xr[t + i * 256];
    vals[i] = v; s += v; sq += v * v;
  }
  for (int off = 16; off; off >>= 1) { s += __shfl_xor(s, off, 32); sq += __shfl_xor(sq, off, 32); }
  __shared__ float rs[8], rq[8];
  int w = t >> 5, l = t & 31;
  if (l == 0) { rs[w] = s; rq[w] = sq; }
  __syncthreads();
  float S = 0.f, Q = 0.f;
#pragma unroll
  for (int i = 0; i < 8; ++i) { S += rs[i]; Q += rq[i]; }
  float mean = S * (1.f / CDIM);
  float var  = Q * (1.f / CDIM) - mean * mean;
  float inv  = rsqrtf(var + 1e-5f);
#pragma unroll
  for (int i = 0; i < 4; ++i) {
    int c = t + i * 256;
    float h = (vals[i] - mean) * inv * g[c] + bt[c];
    h = fminf(fmaxf(h, 0.f), 1.f);
    out[(size_t)row * CDIM + c] = (unsigned char)(int)rintf(h * 15.f);
  }
}

__global__ __launch_bounds__(256) void k_transpose_v(const half_t* __restrict__ vf,
                                                     half_t* __restrict__ vT) {
  __shared__ half_t tile[64][66];
  int bh = blockIdx.y, t0 = blockIdx.x * 64, t = threadIdx.x;
  for (int i = t; i < 2048; i += 256) {
    int r = i >> 5, c2 = i & 31;
    unsigned u = *(const unsigned*)(vf + ((size_t)bh * NPAD + t0 + r) * DHEAD + c2 * 2);
    *(unsigned*)&tile[r][c2 * 2] = u;
  }
  __syncthreads();
  for (int i = t; i < 2048; i += 256) {
    int d = i >> 5, n2 = i & 31;
    half_t h0 = tile[n2 * 2][d], h1 = tile[n2 * 2 + 1][d];
    unsigned u = (unsigned)*(unsigned short*)&h0 | ((unsigned)*(unsigned short*)&h1 << 16);
    *(unsigned*)(vT + ((size_t)bh * DHEAD + d) * NPAD + t0 + n2 * 2) = u;
  }
}

struct GP {
  const unsigned char* A;
  const signed char*   W;
  int M, N, K;
  const float* bias;
  float*       out;
  const float* resid;
  half_t *q, *k, *v;
  unsigned char* aout;
};

template <int EPI>
__global__ __launch_bounds__(256) void k_gemm_q4(GP p) {
  __shared__ unsigned ls[4 * 128 * LROW];
  int t = threadIdx.x, w = t >> 5, l = t & 31, lm = l & 15, lh = l >> 4;
  int n0 = blockIdx.x * 128, m0 = blockIdx.y * 128;

  int cc = t & 3;
  int rr0 = t >> 2, rr1 = 64 + (t >> 2);
  int gmA0 = m0 + rr0; if (gmA0 >= p.M) gmA0 = p.M - 1;
  int gmA1 = m0 + rr1; if (gmA1 >= p.M) gmA1 = p.M - 1;
  int gnB0 = n0 + rr0; if (gnB0 >= p.N) gnB0 = p.N - 1;
  int gnB1 = n0 + rr1; if (gnB1 >= p.N) gnB1 = p.N - 1;

  auto stage = [&](int buf, int k0) {
    unsigned baseA = (unsigned)(buf * 2) * (128 * LROW);
    unsigned baseB = baseA + 128 * LROW;
    unsigned ldsA0 = (unsigned)(unsigned long long)&ls[baseA + rr0 * LROW + cc * 4];
    unsigned ldsA1 = (unsigned)(unsigned long long)&ls[baseA + rr1 * LROW + cc * 4];
    unsigned ldsB0 = (unsigned)(unsigned long long)&ls[baseB + rr0 * LROW + cc * 4];
    unsigned ldsB1 = (unsigned)(unsigned long long)&ls[baseB + rr1 * LROW + cc * 4];
    unsigned offA0 = (unsigned)((size_t)gmA0 * p.K + k0 + cc * 16);
    unsigned offA1 = (unsigned)((size_t)gmA1 * p.K + k0 + cc * 16);
    unsigned offB0 = (unsigned)((size_t)gnB0 * p.K + k0 + cc * 16);
    unsigned offB1 = (unsigned)((size_t)gnB1 * p.K + k0 + cc * 16);
    asm volatile("global_load_async_to_lds_b128 %0, %1, %2"
                 :: "v"(ldsA0), "v"(offA0), "s"(p.A) : "memory");
    asm volatile("global_load_async_to_lds_b128 %0, %1, %2"
                 :: "v"(ldsA1), "v"(offA1), "s"(p.A) : "memory");
    asm volatile("global_load_async_to_lds_b128 %0, %1, %2"
                 :: "v"(ldsB0), "v"(offB0), "s"(p.W) : "memory");
    asm volatile("global_load_async_to_lds_b128 %0, %1, %2"
                 :: "v"(ldsB1), "v"(offB1), "s"(p.W) : "memory");
  };

  v8i acc[4][2];
#pragma unroll
  for (int mt = 0; mt < 4; ++mt)
#pragma unroll
    for (int nt = 0; nt < 2; ++nt) acc[mt][nt] = (v8i){0, 0, 0, 0, 0, 0, 0, 0};

  int T = p.K >> 6;
  stage(0, 0);
  for (int it = 0; it < T; ++it) {
    asm volatile("s_wait_asynccnt 0x0" ::: "memory");
    __syncthreads();
    if (it + 1 < T) stage((it + 1) & 1, (it + 1) << 6);
    if (it + 2 < T) {
      int k2 = (it + 2) << 6;
      int gm = m0 + (t >> 1); if (gm >= p.M) gm = p.M - 1;
      int gn = n0 + (t >> 1); if (gn >= p.N) gn = p.N - 1;
      __builtin_prefetch(p.A + (size_t)gm * p.K + k2 + (t & 1) * 32, 0, 0);
      __builtin_prefetch(p.W + (size_t)gn * p.K + k2 + (t & 1) * 32, 0, 0);
    }
    unsigned baseA = (unsigned)((it & 1) * 2) * (128 * LROW);
    unsigned baseB = baseA + 128 * LROW;

    U8 aF[4], bF[2];
#pragma unroll
    for (int mt = 0; mt < 4; ++mt) {
      int row = (w & 1) * 64 + mt * 16 + lm;
#pragma unroll
      for (int v = 0; v < 8; ++v)
        aF[mt].u[v] = ls[baseA + row * LROW + ((v >> 1) * 4 + lh * 2 + (v & 1))];
    }
#pragma unroll
    for (int nt = 0; nt < 2; ++nt) {
      int row = (w >> 1) * 32 + nt * 16 + lm;
#pragma unroll
      for (int v = 0; v < 8; ++v)
        bF[nt].u[v] = ls[baseB + row * LROW + ((v >> 1) * 4 + lh * 2 + (v & 1))];
    }
#pragma unroll
    for (int mt = 0; mt < 4; ++mt)
#pragma unroll
      for (int nt = 0; nt < 2; ++nt)
        acc[mt][nt] = __builtin_amdgcn_wmma_i32_16x16x64_iu8(
            false, aF[mt].i, true, bF[nt].i, acc[mt][nt], false, false);
  }

  int brow = m0 + (w & 1) * 64, bcol = n0 + (w >> 1) * 32;
#pragma unroll
  for (int mt = 0; mt < 4; ++mt)
#pragma unroll
    for (int nt = 0; nt < 2; ++nt)
#pragma unroll
      for (int v = 0; v < 8; ++v) {
        int m = brow + mt * 16 + lh * 8 + v;
        int n = bcol + nt * 16 + lm;
        if (m >= p.M || n >= p.N) continue;
        float a = (float)acc[mt][nt][v];
        if constexpr (EPI == 0) {
          float qi = rintf(a * (1.f / 15.f));
          int part = n >> 10, hh = (n >> 6) & 15, d = n & 63;
          int bb = m / NSEQ, nn = m % NSEQ;
          half_t* dst = (part == 0) ? p.q : ((part == 1) ? p.k : p.v);
          dst[(((size_t)bb * NHEAD + hh) * NPAD + nn) * DHEAD + d] = (half_t)qi;
        } else if constexpr (EPI == 1) {
          float y  = a * (1.f / 225.f) + p.bias[n];
          float yq = rintf(y * 15.f) * (1.f / 15.f);
          p.out[(size_t)m * CDIM + n] = p.resid[(size_t)m * CDIM + n] + yq;
        } else if constexpr (EPI == 2) {
          float y  = a * (1.f / 225.f) + p.bias[n];
          float yq = rintf(y * 15.f) * (1.f / 15.f);
          float g  = 0.5f * yq * (1.f + erff(yq * 0.70710678118654752f));
          g = fminf(fmaxf(g, 0.f), 1.f);
          p.aout[(size_t)m * HIDDEN + n] = (unsigned char)(int)rintf(g * 15.f);
        } else {
          float y  = a * (1.f / 225.f) + p.bias[n];
          float yq = rintf(y * 15.f) * (1.f / 15.f);
          float y2 = rintf(yq * 7.f) * (1.f / 7.f);
          p.out[(size_t)m * CDIM + n] += y2;
        }
      }
}

__global__ __launch_bounds__(256) void k_attention(const half_t* __restrict__ qf,
                                                   const half_t* __restrict__ kf,
                                                   const half_t* __restrict__ vT,
                                                   unsigned char* __restrict__ aq) {
  __shared__ float    ss[16 * SPAD];
  __shared__ unsigned ps[16 * (SPAD / 2)];
  __shared__ unsigned qs[16 * 32];
  __shared__ float    red[4 * 16 * 16];
  int t = threadIdx.x, w = t >> 5, l = t & 31, lm = l & 15, lh = l >> 4;
  int qt = blockIdx.x, bh = blockIdx.y;

  for (int i = t; i < 16 * SPAD; i += 256) ss[i] = -3.0e38f;
  for (int i = t; i < 512; i += 256) {
    int r = i >> 5, c = i & 31;
    qs[i] = *(const unsigned*)(qf + ((size_t)bh * NPAD + qt * 16 + r) * DHEAD + c * 2);
  }
  __syncthreads();

  for (int kt = w; kt < QTILES; kt += 8) {
    U16 a0, a1, b0, b1;
#pragma unroll
    for (int v = 0; v < 8; ++v) {
      int ci = lh * 8 + v;
      a0.u[v] = qs[lm * 32 + ci];
      a1.u[v] = qs[lm * 32 + 16 + ci];
    }
    const half_t* kr = kf + ((size_t)bh * NPAD + kt * 16 + lm) * DHEAD;
#pragma unroll
    for (int v = 0; v < 8; ++v) {
      b0.u[v] = *(const unsigned*)(kr + lh * 16 + v * 2);
      b1.u[v] = *(const unsigned*)(kr + 32 + lh * 16 + v * 2);
    }
    v8f sc = {0.f, 0.f, 0.f, 0.f, 0.f, 0.f, 0.f, 0.f};
    sc = __builtin_amdgcn_wmma_f32_16x16x32_f16(false, a0.h, false, b0.h, (short)0, sc, false, false);
    sc = __builtin_amdgcn_wmma_f32_16x16x32_f16(false, a1.h, false, b1.h, (short)0, sc, false, false);
#pragma unroll
    for (int v = 0; v < 8; ++v) {
      int m = lh * 8 + v, jj = kt * 16 + lm;
      if (jj < NSEQ) ss[m * SPAD + jj] = sc[v] * (1.f / 1800.f);
    }
  }
  __syncthreads();

  {
    int row = t >> 4, li = t & 15;
    float mx = -3.0e38f;
    for (int j = li; j < SPAD; j += 16) mx = fmaxf(mx, ss[row * SPAD + j]);
    for (int off = 8; off; off >>= 1) mx = fmaxf(mx, __shfl_xor(mx, off, 32));
    float sm = 0.f;
    for (int j = li; j < SPAD; j += 16) {
      float e = expf(ss[row * SPAD + j] - mx);
      ss[row * SPAD + j] = e;
      sm += e;
    }
    for (int off = 8; off; off >>= 1) sm += __shfl_xor(sm, off, 32);
    float inv7 = 7.f / sm;
    __syncthreads();
    for (int j2 = li; j2 < SPAD / 2; j2 += 16) {
      half_t h0 = (half_t)rintf(ss[row * SPAD + j2 * 2]     * inv7);
      half_t h1 = (half_t)rintf(ss[row * SPAD + j2 * 2 + 1] * inv7);
      ps[row * (SPAD / 2) + j2] =
          (unsigned)*(unsigned short*)&h0 | ((unsigned)*(unsigned short*)&h1 << 16);
    }
  }
  __syncthreads();

  int ntile = w & 3, khalf = w >> 2;
  int dcol = ntile * 16 + lm;
  const half_t* vrow = vT + ((size_t)bh * DHEAD + dcol) * NPAD;
  v8f oacc = {0.f, 0.f, 0.f, 0.f, 0.f, 0.f, 0.f, 0.f};
  for (int st = khalf; st < SPAD / 32; st += 2) {
    U16 pa, vb;
#pragma unroll
    for (int v = 0; v < 8; ++v)
      pa.u[v] = ps[lm * (SPAD / 2) + st * 16 + lh * 8 + v];
#pragma unroll
    for (int v = 0; v < 8; ++v)
      vb.u[v] = *(const unsigned*)(vrow + st * 32 + lh * 16 + v * 2);
    oacc = __builtin_amdgcn_wmma_f32_16x16x32_f16(false, pa.h, false, vb.h, (short)0, oacc, false, false);
  }
  if (khalf == 1)
#pragma unroll
    for (int v = 0; v < 8; ++v) red[(ntile * 16 + lh * 8 + v) * 16 + lm] = oacc[v];
  __syncthreads();
  if (khalf == 0) {
    int b = bh >> 4, h = bh & 15;
#pragma unroll
    for (int v = 0; v < 8; ++v) {
      int m = lh * 8 + v, qrow = qt * 16 + m;
      if (qrow < NSEQ) {
        float o = (oacc[v] + red[(ntile * 16 + m) * 16 + lm]) * (1.f / 105.f);
        o = fminf(fmaxf(o, 0.f), 1.f);
        aq[((size_t)b * NSEQ + qrow) * CDIM + h * DHEAD + dcol] = (unsigned char)(int)rintf(o * 15.f);
      }
    }
  }
}

extern "C" void kernel_launch(void* const* d_in, const int* in_sizes, int n_in,
                              void* d_out, int out_size, void* d_ws, size_t ws_size,
                              hipStream_t stream) {
  const float* x      = (const float*)d_in[0];
  const float* ln1_w  = (const float*)d_in[1];
  const float* ln1_b  = (const float*)d_in[2];
  const float* qkv_w  = (const float*)d_in[3];
  const float* proj_w = (const float*)d_in[4];
  const float* proj_b = (const float*)d_in[5];
  const float* ln2_w  = (const float*)d_in[6];
  const float* ln2_b  = (const float*)d_in[7];
  const float* fc1_w  = (const float*)d_in[8];
  const float* fc1_b  = (const float*)d_in[9];
  const float* fc2_w  = (const float*)d_in[10];
  const float* fc2_b  = (const float*)d_in[11];
  float* out = (float*)d_out;

  char* base = (char*)d_ws;
  size_t off = 0;
  auto carve = [&](size_t sz) -> void* {
    void* p = base + off;
    off += (sz + 255) & ~(size_t)255;
    return p;
  };
  const size_t QKV_BYTES = (size_t)BDIM * NHEAD * NPAD * DHEAD * sizeof(half_t);
  unsigned*      wmax  = (unsigned*)carve(4 * sizeof(unsigned));
  signed char*   qkvq  = (signed char*)carve((size_t)3 * CDIM * CDIM);
  signed char*   projq = (signed char*)carve((size_t)CDIM * CDIM);
  signed char*   fc1q  = (signed char*)carve((size_t)HIDDEN * CDIM);
  signed char*   fc2q  = (signed char*)carve((size_t)CDIM * HIDDEN);
  unsigned char* aq    = (unsigned char*)carve((size_t)MROWS * CDIM);
  unsigned char* h1q   = (unsigned char*)carve((size_t)MROWS * HIDDEN);
  half_t* qf = (half_t*)carve(QKV_BYTES);
  half_t* kf = (half_t*)carve(QKV_BYTES);
  half_t* vf = (half_t*)carve(QKV_BYTES);
  half_t* vT = (half_t*)carve(QKV_BYTES);

  k_zero4<<<1, 256, 0, stream>>>(wmax);
  k_absmax_tanh<<<1024, 256, 0, stream>>>(qkv_w,  3 * CDIM * CDIM, wmax + 0);
  k_absmax_tanh<<<1024, 256, 0, stream>>>(proj_w, CDIM * CDIM,     wmax + 1);
  k_absmax_tanh<<<1024, 256, 0, stream>>>(fc1_w,  HIDDEN * CDIM,   wmax + 2);
  k_absmax_tanh<<<1024, 256, 0, stream>>>(fc2_w,  CDIM * HIDDEN,   wmax + 3);
  k_wquant<<<2048, 256, 0, stream>>>(qkv_w,  3 * CDIM * CDIM, wmax + 0, qkvq);
  k_wquant<<<2048, 256, 0, stream>>>(proj_w, CDIM * CDIM,     wmax + 1, projq);
  k_wquant<<<2048, 256, 0, stream>>>(fc1_w,  HIDDEN * CDIM,   wmax + 2, fc1q);
  k_wquant<<<2048, 256, 0, stream>>>(fc2_w,  CDIM * HIDDEN,   wmax + 3, fc2q);

  k_zerobuf<<<4096, 256, 0, stream>>>((uint4*)qf, 3 * QKV_BYTES / 16);

  int mtiles = (MROWS + 127) / 128;

  k_ln_quant<<<MROWS, 256, 0, stream>>>(x, ln1_w, ln1_b, aq);
  {
    GP p = {};
    p.A = aq; p.W = qkvq; p.M = MROWS; p.N = 3 * CDIM; p.K = CDIM;
    p.q = qf; p.k = kf; p.v = vf;
    k_gemm_q4<0><<<dim3(3 * CDIM / 128, mtiles), 256, 0, stream>>>(p);
  }
  k_transpose_v<<<dim3(NPAD / 64, BDIM * NHEAD), 256, 0, stream>>>(vf, vT);
  k_attention<<<dim3(QTILES, BDIM * NHEAD), 256, 0, stream>>>(qf, kf, vT, aq);
  {
    GP p = {};
    p.A = aq; p.W = projq; p.M = MROWS; p.N = CDIM; p.K = CDIM;
    p.bias = proj_b; p.out = out; p.resid = x;
    k_gemm_q4<1><<<dim3(CDIM / 128, mtiles), 256, 0, stream>>>(p);
  }

  k_ln_quant<<<MROWS, 256, 0, stream>>>(out, ln2_w, ln2_b, aq);
  {
    GP p = {};
    p.A = aq; p.W = fc1q; p.M = MROWS; p.N = HIDDEN; p.K = CDIM;
    p.bias = fc1_b; p.aout = h1q;
    k_gemm_q4<2><<<dim3(HIDDEN / 128, mtiles), 256, 0, stream>>>(p);
  }
  {
    GP p = {};
    p.A = h1q; p.W = fc2q; p.M = MROWS; p.N = CDIM; p.K = HIDDEN;
    p.bias = fc2_b; p.out = out;
    k_gemm_q4<3><<<dim3(CDIM / 128, mtiles), 256, 0, stream>>>(p);
  }
}
